// UIA_O_Model_51642686767395
// MI455X (gfx1250) — matrix operand order repaired and checked by reading, not compiled, not run on hardware
//
#include <hip/hip_runtime.h>
#include <math.h>

typedef __attribute__((ext_vector_type(16))) __bf16 v16bf;
typedef __attribute__((ext_vector_type(8)))  float  v8f;
typedef long long i64;

static constexpr int Bsz  = 8192;
static constexpr int DIN  = 1024;
static constexpr int H    = 2048;
static constexpr int DOUT = 1024;
static constexpr int P    = 4;
static constexpr int E    = 6;
static constexpr int C    = 512;   // H/P
static constexpr int C2   = 256;   // C/2

// ---------- helpers ----------
__device__ __forceinline__ unsigned short f2bf(float f) {
    unsigned u = __float_as_uint(f);
    unsigned r = u + 0x7FFFu + ((u >> 16) & 1u);
    return (unsigned short)(r >> 16);
}

union FragU { v16bf v; uint4 q[2]; };

// 16B per-lane async copy global -> LDS (gfx1250, ASYNCcnt-tracked)
__device__ __forceinline__ void async_cp16(unsigned ldsByteAddr, const unsigned short* g) {
    asm volatile("global_load_async_to_lds_b128 %0, %1, off"
                 :: "v"(ldsByteAddr), "v"(g) : "memory");
}
__device__ __forceinline__ void wait_async_le8() { asm volatile("s_wait_asynccnt 0x8" ::: "memory"); }
__device__ __forceinline__ void wait_async_0()   { asm volatile("s_wait_asynccnt 0x0" ::: "memory"); }

// ---------- generic bf16 GEMM, f32 accumulate via v_wmma_f32_16x16x32_bf16 ----------
// out[M,N] = A[M,K]*B[K,N] (+bias[N]); B given TRANSPOSED: BT[N,K] (K contiguous, ldb >= K extent)
// Tiles 128x128x64, 256 threads = 8 waves; wave w owns rows [16w,16w+16) x 128 cols.
// Double-buffered LDS filled by async copies; stage s+1 issued before computing stage s.
// MODE: 0 = store f32, 1 = accumulate into f32, 2 = store bf16
template<int MODE>
__global__ void __launch_bounds__(256)
gemm_bf16_kernel(const unsigned short* __restrict__ A, int lda,
                 const unsigned short* __restrict__ BT, int ldb,
                 const float* __restrict__ bias,
                 void* __restrict__ outp, int ldo, int K)
{
    // [buffer][A/B][128 rows x 64 k, stride 72]
    __shared__ __align__(16) unsigned short ldsbuf[2][2][128 * 72];

    const int tid  = threadIdx.x;
    const int lane = tid & 31;
    const int wave = tid >> 5;
    const int bm   = blockIdx.y * 128;
    const int bn   = blockIdx.x * 128;

    v8f acc[8];
    const v8f vz = {0.f, 0.f, 0.f, 0.f, 0.f, 0.f, 0.f, 0.f};
#pragma unroll
    for (int i = 0; i < 8; ++i) acc[i] = vz;

    // staging: thread t copies 64B of A row (t>>1) and 64B of BT row (t>>1)
    const int srow = tid >> 1;          // 0..127
    const int scol = (tid & 1) * 32;    // 0 or 32

    const unsigned ldsBase     = (unsigned)(size_t)&ldsbuf[0][0][0];
    const unsigned ldsStageOff = (unsigned)((srow * 72 + scol) * 2);

    const unsigned short* ga = A  + (i64)(bm + srow) * lda + scol;
    const unsigned short* gb = BT + (i64)(bn + srow) * ldb + scol;

    const int m   = lane & 15;
    const int khi = lane >> 4;          // 0 or 1

    auto issue = [&](int buf) {
        const unsigned la = ldsBase + (unsigned)buf * 36864u + ldsStageOff;
        const unsigned lb = la + 18432u;
        async_cp16(la,      ga);
        async_cp16(la + 16, ga + 8);
        async_cp16(la + 32, ga + 16);
        async_cp16(la + 48, ga + 24);
        async_cp16(lb,      gb);
        async_cp16(lb + 16, gb + 8);
        async_cp16(lb + 32, gb + 16);
        async_cp16(lb + 48, gb + 24);
        ga += 64;
        gb += 64;
    };

    auto compute = [&](const unsigned short* lA, const unsigned short* lB) {
#pragma unroll
        for (int kc = 0; kc < 2; ++kc) {
            const int kb = kc * 32;
            FragU af;
            {
                const int ra = (16 * wave + m) * 72 + kb;
                af.q[0] = *reinterpret_cast<const uint4*>(&lA[ra + khi * 16]);
                af.q[1] = *reinterpret_cast<const uint4*>(&lA[ra + khi * 16 + 8]);
            }
            FragU bf[8];
#pragma unroll
            for (int nt = 0; nt < 8; ++nt) {
                const int rb = (nt * 16 + m) * 72 + kb + khi * 16;
                bf[nt].q[0] = *reinterpret_cast<const uint4*>(&lB[rb]);
                bf[nt].q[1] = *reinterpret_cast<const uint4*>(&lB[rb + 8]);
            }
#pragma unroll
            for (int nt = 0; nt < 8; ++nt)
                acc[nt] = __builtin_amdgcn_wmma_f32_16x16x32_bf16(
                    false, af.v, false, bf[nt].v, (short)0, acc[nt], false, false);
        }
    };

    const int nst = K >> 6;   // K is a multiple of 64 for all call sites
    issue(0);
    for (int s = 0; s < nst - 1; ++s) {
        issue((s + 1) & 1);
        wait_async_le8();
        __syncthreads();
        compute(&ldsbuf[s & 1][0][0], &ldsbuf[s & 1][1][0]);
        __syncthreads();
    }
    wait_async_0();
    __syncthreads();
    compute(&ldsbuf[(nst - 1) & 1][0][0], &ldsbuf[(nst - 1) & 1][1][0]);

    // epilogue: row = bm + 16*wave + r + 8*khi, col = bn + nt*16 + m
#pragma unroll
    for (int nt = 0; nt < 8; ++nt) {
        const int col = bn + nt * 16 + m;
        const float bv = bias ? bias[col] : 0.f;
#pragma unroll
        for (int r = 0; r < 8; ++r) {
            const int row = bm + 16 * wave + r + 8 * khi;
            float v = acc[nt][r] + bv;
            const i64 idx = (i64)row * ldo + col;
            if constexpr (MODE == 2) {
                ((unsigned short*)outp)[idx] = f2bf(v);
            } else {
                float* o = (float*)outp;
                if constexpr (MODE == 1) v += o[idx];
                o[idx] = v;
            }
        }
    }
}

// ---------- conversion kernels ----------
__global__ void convert_f32_bf16(const float* __restrict__ src, unsigned short* __restrict__ dst, i64 n) {
    for (i64 i = (i64)blockIdx.x * blockDim.x + threadIdx.x; i < n; i += (i64)gridDim.x * blockDim.x)
        dst[i] = f2bf(src[i]);
}

// batched transpose + convert: dst[b][n][k] = bf16(src[b][k][n]); K,N multiples of 32
__global__ void convT_kernel(const float* __restrict__ src, unsigned short* __restrict__ dst, int K, int N) {
    __shared__ float tile[32][33];
    const i64 bofs = (i64)blockIdx.z * K * N;
    const int k0 = blockIdx.y * 32, n0 = blockIdx.x * 32;
#pragma unroll
    for (int r = threadIdx.y; r < 32; r += 8)
        tile[r][threadIdx.x] = src[bofs + (i64)(k0 + r) * N + (n0 + threadIdx.x)];
    __syncthreads();
#pragma unroll
    for (int r = threadIdx.y; r < 32; r += 8)
        dst[bofs + (i64)(n0 + r) * K + (k0 + threadIdx.x)] = f2bf(tile[threadIdx.x][r]);
}

__global__ void zero_f32(float* __restrict__ dst, i64 n) {
    for (i64 i = (i64)blockIdx.x * blockDim.x + threadIdx.x; i < n; i += (i64)gridDim.x * blockDim.x)
        dst[i] = 0.f;
}

// ---------- elementwise / reduction kernels ----------
__global__ void diffuse_kernel(const float* __restrict__ h, const float* __restrict__ inc,
                               const float* __restrict__ damping,
                               float* __restrict__ difF, unsigned short* __restrict__ difB,
                               unsigned short* __restrict__ baseB)
{
    const i64 idx = (i64)blockIdx.x * blockDim.x + threadIdx.x;
    if (idx >= (i64)Bsz * C) return;
    const int b = (int)(idx / C), c = (int)(idx % C);

    float L[P][P];
#pragma unroll
    for (int p = 0; p < P; ++p)
#pragma unroll
        for (int q = 0; q < P; ++q) {
            float s = 0.f;
#pragma unroll
            for (int e = 0; e < E; ++e) s += inc[e * P + p] * inc[e * P + q];
            L[p][q] = s;
        }
    const float dmp = damping[0];

    float hv[P];
#pragma unroll
    for (int p = 0; p < P; ++p) hv[p] = h[(i64)b * H + p * C + c];

    float bsum = 0.f;
#pragma unroll
    for (int p = 0; p < P; ++p) {
        float s = 0.f;
#pragma unroll
        for (int q = 0; q < P; ++q) s += L[p][q] * hv[q];
        const float d = hv[p] - dmp * s;
        const i64 o = (i64)b * H + p * C + c;
        difF[o] = d;
        difB[o] = f2bf(d);
        bsum += d;
    }
    baseB[idx] = f2bf(bsum * 0.25f);
}

__global__ void weighted_kernel(const float* __restrict__ h, const float* __restrict__ inc,
                                int e, unsigned short* __restrict__ out)
{
    const i64 idx = (i64)blockIdx.x * blockDim.x + threadIdx.x;
    if (idx >= (i64)Bsz * C) return;
    const int b = (int)(idx / C), c = (int)(idx % C);
    float w = 0.f;
#pragma unroll
    for (int p = 0; p < P; ++p) w += inc[e * P + p] * h[(i64)b * H + p * C + c];
    out[idx] = f2bf(w);
}

__global__ void diffpair_kernel(const float* __restrict__ dif, int ii, int jj,
                                unsigned short* __restrict__ out)
{
    const i64 idx = (i64)blockIdx.x * blockDim.x + threadIdx.x;
    if (idx >= (i64)Bsz * C) return;
    const int b = (int)(idx / C), c = (int)(idx % C);
    out[idx] = f2bf(dif[(i64)b * H + ii * C + c] - dif[(i64)b * H + jj * C + c]);
}

__global__ void normacc_kernel(const float* __restrict__ T, float* __restrict__ h1n, float scale)
{
    __shared__ float red[256];
    const int b = blockIdx.x;
    float s = 0.f;
    for (int t = threadIdx.x; t < C; t += 256) {
        const float v = T[(i64)b * C + t];
        s += v * v;
    }
    red[threadIdx.x] = s;
    __syncthreads();
    for (int st = 128; st > 0; st >>= 1) {
        if (threadIdx.x < st) red[threadIdx.x] += red[threadIdx.x + st];
        __syncthreads();
    }
    if (threadIdx.x == 0) h1n[b] += sqrtf(red[0]) * scale;
}

__global__ void acc_total_kernel(const float* __restrict__ S, float* __restrict__ total, i64 n)
{
    for (i64 i = (i64)blockIdx.x * blockDim.x + threadIdx.x; i < n; i += (i64)gridDim.x * blockDim.x)
        total[i] += 0.25f * S[i];
}

__global__ void sqsum_kernel(const float* __restrict__ T, i64 n, float* __restrict__ loss, float scale)
{
    __shared__ float red[256];
    float s = 0.f;
    for (i64 i = (i64)blockIdx.x * blockDim.x + threadIdx.x; i < n; i += (i64)gridDim.x * blockDim.x) {
        const float v = T[i];
        s += v * v;
    }
    red[threadIdx.x] = s;
    __syncthreads();
    for (int st = 128; st > 0; st >>= 1) {
        if (threadIdx.x < st) red[threadIdx.x] += red[threadIdx.x + st];
        __syncthreads();
    }
    if (threadIdx.x == 0) atomicAdd(loss, red[0] * scale);
}

// ---------- host launch ----------
template<int MODE>
static void gemm(hipStream_t st, const unsigned short* A, int lda, const unsigned short* BT, int ldb,
                 const float* bias, void* out, int ldo, int M, int N, int K)
{
    dim3 g(N / 128, M / 128);
    gemm_bf16_kernel<MODE><<<g, 256, 0, st>>>(A, lda, BT, ldb, bias, out, ldo, K);
}

static void conv(hipStream_t s, const float* src, unsigned short* dst, i64 n) {
    int blocks = (int)((n + 1023) / 1024);
    if (blocks > 8192) blocks = 8192;
    convert_f32_bf16<<<blocks, 256, 0, s>>>(src, dst, n);
}

static void convT(hipStream_t s, const float* src, unsigned short* dst, int batch, int K, int N) {
    dim3 g(N / 32, K / 32, batch);
    convT_kernel<<<g, dim3(32, 8), 0, s>>>(src, dst, K, N);
}

extern "C" void kernel_launch(void* const* d_in, const int* in_sizes, int n_in,
                              void* d_out, int out_size, void* d_ws, size_t ws_size,
                              hipStream_t stream)
{
    (void)in_sizes; (void)n_in; (void)out_size; (void)ws_size;

    const float* x     = (const float*)d_in[0];
    const float* fiber = (const float*)d_in[1];
    const float* W_in  = (const float*)d_in[2];
    const float* b_in  = (const float*)d_in[3];
    const float* inc   = (const float*)d_in[4];
    const float* sheaf = (const float*)d_in[5];
    const float* damp  = (const float*)d_in[6];
    const float* Wb    = (const float*)d_in[7];
    const float* bb    = (const float*)d_in[8];
    const float* Wf    = (const float*)d_in[9];
    const float* bfv   = (const float*)d_in[10];
    const float* Wt    = (const float*)d_in[11];
    const float* bt    = (const float*)d_in[12];
    const float* Wr    = (const float*)d_in[13];
    /* br cancels in (ri-rj) */
    const float* Wg    = (const float*)d_in[15];
    const float* bg    = (const float*)d_in[16];
    const float* Wc    = (const float*)d_in[17];
    const float* bc    = (const float*)d_in[18];

    float* out = (float*)d_out;
    const i64 OFF_OUT = 0;
    const i64 OFF_DIF = OFF_OUT + (i64)Bsz * DOUT;
    const i64 OFF_H1N = OFF_DIF + (i64)Bsz * H;
    const i64 OFF_H1L = OFF_H1N + Bsz;
    const i64 OFF_TOT = OFF_H1L + 1;

    // workspace carve-up
    char* w = (char*)d_ws;
    auto take = [&](size_t bytes) { char* p = w; w += (bytes + 255) & ~(size_t)255; return p; };
    float*          h_f32   = (float*)         take((size_t)Bsz * H * 4);
    unsigned short* x_bf    = (unsigned short*)take((size_t)Bsz * DIN * 2);
    unsigned short* winT    = (unsigned short*)take((size_t)DIN * H * 2);       // [H][DIN]
    unsigned short* dif_bf  = (unsigned short*)take((size_t)Bsz * H * 2);
    unsigned short* glue_bf = (unsigned short*)take((size_t)Bsz * H * 2);
    unsigned short* base_bf = (unsigned short*)take((size_t)Bsz * C * 2);       // later reused as total_bf
    unsigned short* fib_bf  = (unsigned short*)take((size_t)Bsz * C * 2);
    unsigned short* tmp_bf  = (unsigned short*)take((size_t)Bsz * C * 2);
    float*          tmp_f   = (float*)         take((size_t)Bsz * C * 4);
    unsigned short* sheafT  = (unsigned short*)take((size_t)E * C * C * 2);     // per e: [C][C]
    unsigned short* wbT     = (unsigned short*)take((size_t)P * C * C2 * 2);    // per s: [C2][C]
    unsigned short* wfT     = (unsigned short*)take((size_t)P * C * C2 * 2);
    unsigned short* wtT     = (unsigned short*)take((size_t)C * C * 2);         // [C][C]
    unsigned short* wrT     = (unsigned short*)take((size_t)E * C * C2 * 2);    // per e: [C2][C]
    unsigned short* wgT     = (unsigned short*)take((size_t)H * H * 2);         // [H][H]
    unsigned short* wcT     = (unsigned short*)take((size_t)(H + C) * DOUT * 2);// [DOUT][H+C]

    // 1) conversions (A-side plain, B-side transposed)
    conv (stream, x,     x_bf,  (i64)Bsz * DIN);
    conv (stream, fiber, fib_bf,(i64)Bsz * C);
    convT(stream, W_in,  winT,  1, DIN,   H);
    convT(stream, sheaf, sheafT,E, C,     C);
    convT(stream, Wb,    wbT,   P, C,     C2);
    convT(stream, Wf,    wfT,   P, C,     C2);
    convT(stream, Wt,    wtT,   1, C,     C);
    convT(stream, Wr,    wrT,   E, C,     C2);
    convT(stream, Wg,    wgT,   1, H,     H);
    convT(stream, Wc,    wcT,   1, H + C, DOUT);

    // 2) h = x @ W_in + b_in
    gemm<0>(stream, x_bf, DIN, winT, DIN, b_in, h_f32, H, Bsz, H, DIN);

    // 3) diffused (-> d_out), dif_bf, base_bf
    {
        const int nb = (int)(((i64)Bsz * C + 255) / 256);
        diffuse_kernel<<<nb, 256, 0, stream>>>(h_f32, inc, damp, out + OFF_DIF, dif_bf, base_bf);
    }

    // 4) zero h1_norm + h1_loss + total (contiguous in d_out)
    zero_f32<<<8192, 256, 0, stream>>>(out + OFF_H1N, (i64)(Bsz + 1 + (i64)Bsz * C));

    // 5) h1_norm: per edge, coboundary_e = weighted_e @ sheaf_e
    for (int e = 0; e < E; ++e) {
        const int nb = (int)(((i64)Bsz * C + 255) / 256);
        weighted_kernel<<<nb, 256, 0, stream>>>(h_f32, inc, e, tmp_bf);
        gemm<0>(stream, tmp_bf, C, sheafT + (i64)e * C * C, C, nullptr, tmp_f, C, Bsz, C, C);
        normacc_kernel<<<Bsz, 256, 0, stream>>>(tmp_f, out + OFF_H1N, 1.0f / E);
    }

    // 6) simplices: U_s = [base@Wb_s+bb_s ; fiber@Wf_s+bf_s]; S_s = U_s@Wt+bt; total = mean_s S_s
    for (int s = 0; s < P; ++s) {
        gemm<2>(stream, base_bf, C, wbT + (i64)s * C * C2, C, bb + s * C2,  tmp_bf,      C, Bsz, C2, C);
        gemm<2>(stream, fib_bf,  C, wfT + (i64)s * C * C2, C, bfv + s * C2, tmp_bf + C2, C, Bsz, C2, C);
        gemm<0>(stream, tmp_bf, C, wtT, C, bt, tmp_f, C, Bsz, C, C);
        acc_total_kernel<<<8192, 256, 0, stream>>>(tmp_f, out + OFF_TOT, (i64)Bsz * C);
    }
    conv(stream, out + OFF_TOT, base_bf, (i64)Bsz * C);  // total -> bf16 (reuse base_bf)

    // 7) h1_loss: per edge, (s_i - s_j) @ Wr_e (bias cancels), mean of squares
    {
        static const int II[E] = {0, 0, 0, 1, 1, 2};
        static const int JJ[E] = {1, 2, 3, 2, 3, 3};
        const float scale = 1.0f / ((float)Bsz * (float)C2 * (float)E);
        for (int e = 0; e < E; ++e) {
            const int nb = (int)(((i64)Bsz * C + 255) / 256);
            diffpair_kernel<<<nb, 256, 0, stream>>>(out + OFF_DIF, II[e], JJ[e], tmp_bf);
            gemm<0>(stream, tmp_bf, C, wrT + (i64)e * C * C2, C, nullptr, tmp_f, C2, Bsz, C2, C);
            sqsum_kernel<<<2048, 256, 0, stream>>>(tmp_f, (i64)Bsz * C2, out + OFF_H1L, scale);
        }
    }

    // 8) glued = diffused @ Wg + bg (bf16 out only)
    gemm<2>(stream, dif_bf, H, wgT, H, bg, glue_bf, H, Bsz, H, H);

    // 9) output = glued @ Wc[:H] + total @ Wc[H:] + bc
    gemm<0>(stream, glue_bf, H, wcT,     H + C, bc,      out + OFF_OUT, DOUT, Bsz, DOUT, H);
    gemm<1>(stream, base_bf, C, wcT + H, H + C, nullptr, out + OFF_OUT, DOUT, Bsz, DOUT, C);
}
